// LatSim_56848187129859
// MI455X (gfx1250) — matrix operand order repaired and checked by reading, not compiled, not run on hardware
//
#include <hip/hip_runtime.h>
#include <hip/hip_bf16.h>

typedef __attribute__((ext_vector_type(16))) _Float16 v16h;
typedef __attribute__((ext_vector_type(8)))  float    v8f;

#define D      512
#define NTR    16384
#define NT     8192
#define NCLS   16
#define TILE_M 16
#define CHUNK  32
#define KSPLIT 4   // waves per block, each covering NTR/KSPLIT keys

// ---------------------------------------------------------------------------
// Tiny init: zero the k-norm-max accumulator (ws is not re-poisoned between
// replays, so this must run every call for determinism).
// ---------------------------------------------------------------------------
__global__ void LatSim_init(unsigned* kmaxbits) {
    if (threadIdx.x == 0 && blockIdx.x == 0) *kmaxbits = 0u;
}

// ---------------------------------------------------------------------------
// One-time prep: transposed f16 copy of ytr -> yT[n][j], n<16, j<16384.
// 512 KB, stays resident in the 192 MB L2 for the whole attention kernel.
// ---------------------------------------------------------------------------
__global__ __launch_bounds__(256) void LatSim_transposeY(
    const float* __restrict__ ytr, _Float16* __restrict__ yT)
{
    const int t = blockIdx.x * 256 + threadIdx.x;   // [0, NCLS*NTR)
    const int n = t >> 14;                          // / NTR
    const int j = t & (NTR - 1);
    yT[t] = (_Float16)ytr[(size_t)j * NCLS + n];
}

// ---------------------------------------------------------------------------
// Low-rank projection: out[row][0..1] = x[row,:] @ A   (A is D x 2).
// One wave per row. Optionally accumulates max ||k_row|| (monotone uint bits).
// ---------------------------------------------------------------------------
__global__ __launch_bounds__(256) void LatSim_proj(
    const float* __restrict__ x, const float* __restrict__ A,
    float* __restrict__ out, int nrows, unsigned* normmax)
{
    const int lane = threadIdx.x & 31;
    const int wave = blockIdx.x * (blockDim.x >> 5) + (threadIdx.x >> 5);
    if (wave >= nrows) return;

    const float* row = x + (size_t)wave * D;
    float a0 = 0.f, a1 = 0.f;
#pragma unroll
    for (int t = 0; t < 4; ++t) {
        const int idx = lane * 4 + 128 * t;
        const float4 xv = *(const float4*)(row + idx);
        const float2 A0 = *(const float2*)(A + 2 * (idx + 0));
        const float2 A1 = *(const float2*)(A + 2 * (idx + 1));
        const float2 A2 = *(const float2*)(A + 2 * (idx + 2));
        const float2 A3 = *(const float2*)(A + 2 * (idx + 3));
        a0 = fmaf(xv.x, A0.x, fmaf(xv.y, A1.x, fmaf(xv.z, A2.x, fmaf(xv.w, A3.x, a0))));
        a1 = fmaf(xv.x, A0.y, fmaf(xv.y, A1.y, fmaf(xv.z, A2.y, fmaf(xv.w, A3.y, a1))));
    }
#pragma unroll
    for (int off = 16; off > 0; off >>= 1) {
        a0 += __shfl_down(a0, off, 32);
        a1 += __shfl_down(a1, off, 32);
    }
    if (lane == 0) {
        out[2 * wave + 0] = a0;
        out[2 * wave + 1] = a1;
        if (normmax != nullptr) {
            // norms >= 0 -> float bits are monotone as unsigned
            atomicMax(normmax, __float_as_uint(sqrtf(a0 * a0 + a1 * a1)));
        }
    }
}

// ---------------------------------------------------------------------------
// Flash-attention over rank-2 logits. Each block owns one 16-row query tile;
// KSPLIT waves split the 16384-key range and combine exact partial sums.
// Row "max" is the upper bound ||q||*max||k|| (softmax-invariant, exp<=1).
// ---------------------------------------------------------------------------
__global__ __launch_bounds__(KSPLIT * 32) void LatSim_attn(
    const float* __restrict__ q, const float* __restrict__ k,
    const _Float16* __restrict__ yT, const unsigned* __restrict__ kmaxbits,
    float* __restrict__ out)
{
    __shared__ float lk[KSPLIT][CHUNK * 2];     // per-wave staged k chunk
    __shared__ float cbuf[KSPLIT - 1][32][8];   // partial C tiles (waves 1..3)
    __shared__ float dbuf[KSPLIT][32];          // partial denominators

    const int lane = threadIdx.x & 31;
    const int wave = threadIdx.x >> 5;
    const int half = lane >> 4;          // 0: lanes 0-15, 1: lanes 16-31
    const int r    = lane & 15;          // query row (A,s) / output column (B,C)
    const int qbase = blockIdx.x * TILE_M;

    const float q0 = q[2 * (qbase + r) + 0];
    const float q1 = q[2 * (qbase + r) + 1];
    const float kmax = __uint_as_float(*kmaxbits);
    const float nm = -sqrtf(q0 * q0 + q1 * q1) * kmax;   // -(row max bound)

    // WMMA A 16x32 f16 K-map: lanes 0-15 -> K {0..15};
    //                         lanes 16-31 -> K {16..31}
    int kA[16];
#pragma unroll
    for (int i = 0; i < 16; ++i)
        kA[i] = 16 * half + i;

    v8f  c = {};
    float dsum = 0.f;
    const int kbeg = wave * (NTR / KSPLIT);
    const int kend = kbeg + (NTR / KSPLIT);

    for (int base = kbeg; base < kend; base += CHUNK) {
        // Stage 32 k-pairs (wave-synchronous LDS: in-order DS, no barrier).
        const float2 kp = *(const float2*)(k + 2 * (size_t)(base + lane));
        lk[wave][2 * lane + 0] = kp.x;
        lk[wave][2 * lane + 1] = kp.y;

        // WMMA B 32x16 f16: lane's 16 halves are contiguous in transposed yT
        // (column n=r, K rows 16*half .. 16*half+15) -> 2x global_load_b128.
        const v16h b = *(const v16h*)(yT + (size_t)r * NTR + base + 16 * half);
        if (base + CHUNK < kend)
            __builtin_prefetch(yT + (size_t)r * NTR + (base + CHUNK) + 16 * half, 0, 3);

        v16h a;
#pragma unroll
        for (int i = 0; i < 16; ++i) {
            const int ja = kA[i];
            // p = exp(q.k - m): fold -m into the fma chain, no extra subtract
            const float p = __expf(
                fmaf(q0, lk[wave][2 * ja], fmaf(q1, lk[wave][2 * ja + 1], nm)));
            dsum += p;
            a[i] = (_Float16)p;
        }
        c = __builtin_amdgcn_wmma_f32_16x16x32_f16(
                /*neg_a=*/false, a, /*neg_b=*/false, b,
                /*c_mod=*/(short)0, c, /*reuse_a=*/false, /*reuse_b=*/false);
    }

    // Combine the KSPLIT partial results (exact: same m bound for all waves).
    if (wave > 0) {
#pragma unroll
        for (int v = 0; v < 8; ++v) cbuf[wave - 1][lane][v] = c[v];
    }
    dbuf[wave][lane] = dsum;
    __syncthreads();

    if (wave == 0) {
#pragma unroll
        for (int w = 0; w < KSPLIT - 1; ++w)
#pragma unroll
            for (int v = 0; v < 8; ++v) c[v] += cbuf[w][lane][v];

        float dtot = dbuf[0][lane] + dbuf[1][lane] + dbuf[2][lane] + dbuf[3][lane];
        dtot += __shfl_xor(dtot, 16, 32);    // lanes l, l^16 jointly cover row r

        // C/D layout: VGPR v holds row (v + 8*half), col = lane&15.
#pragma unroll
        for (int v = 0; v < 8; ++v) {
            const int row = v + 8 * half;
            const float dn = __shfl(dtot, row, 32);   // denom lives at lane==row
            out[(size_t)(qbase + row) * NCLS + r] = c[v] / dn;
        }
    }
}

// ---------------------------------------------------------------------------
extern "C" void kernel_launch(void* const* d_in, const int* in_sizes, int n_in,
                              void* d_out, int out_size, void* d_ws, size_t ws_size,
                              hipStream_t stream) {
    const float* xtr = (const float*)d_in[0];   // (NTR, D)
    const float* ytr = (const float*)d_in[1];   // (NTR, NCLS)
    const float* xt  = (const float*)d_in[2];   // (NT, D)
    const float* A   = (const float*)d_in[3];   // (D, 2)
    float* out = (float*)d_out;                 // (NT, NCLS)

    // Workspace layout (bytes): q 64K | k 128K | kmax 4 | pad | yT 512K
    float*    qbuf = (float*)d_ws;                          // NT*2
    float*    kbuf = qbuf + (size_t)NT * 2;                 // NTR*2
    unsigned* kmaxbits = (unsigned*)(kbuf + (size_t)NTR * 2);
    _Float16* yT = (_Float16*)((char*)d_ws +
                   (((size_t)NT * 2 + (size_t)NTR * 2) * 4 + 64));

    LatSim_init<<<1, 64, 0, stream>>>(kmaxbits);
    LatSim_transposeY<<<(NTR * NCLS) / 256, 256, 0, stream>>>(ytr, yT);
    LatSim_proj<<<NT  / 8, 256, 0, stream>>>(xt,  A, qbuf, NT,  nullptr);
    LatSim_proj<<<NTR / 8, 256, 0, stream>>>(xtr, A, kbuf, NTR, kmaxbits);
    LatSim_attn<<<NT / TILE_M, KSPLIT * 32, 0, stream>>>(qbuf, kbuf, yT, kmaxbits, out);
}
